// GATBranch_52218212385019
// MI455X (gfx1250) — matrix operand order repaired and checked by reading, not compiled, not run on hardware
//
#include <hip/hip_runtime.h>

// ---------------- problem constants (fixed by setup_inputs) ----------------
#define MGR  16       // B*T
#define NGR  5000     // nodes
#define DIN  64       // input feature dim
#define EED  80000    // edges (without self loops)
#define E2   85000    // edges + self loops
#define HH   2        // heads
#define CC   64       // channels per head
#define HC   128      // H*C
#define NEG_SLOPE 0.2f
#define EPSV 1e-16f

typedef __attribute__((ext_vector_type(16))) _Float16 v16h;
typedef __attribute__((ext_vector_type(8)))  float    v8f;

// ---------------- small utility kernels ----------------
__global__ void fill_f32(float* p, float v, long n) {
    long i = (long)blockIdx.x * blockDim.x + threadIdx.x;
    if (i < n) p[i] = v;
}

// ---------------- self-loop weight: mean of incoming edge weights ----------------
__global__ void selfloop_accum(const int* __restrict__ dst, const float* __restrict__ ew,
                               float* __restrict__ cnt, float* __restrict__ wsum, int e_total) {
    int e = blockIdx.x * blockDim.x + threadIdx.x;
    if (e < e_total) {
        int d = dst[e];
        atomicAdd(&cnt[d], 1.0f);
        atomicAdd(&wsum[d], ew[e]);
    }
}

__global__ void selfloop_final(float* __restrict__ selfw, const float* __restrict__ cnt,
                               const float* __restrict__ wsum, int n) {
    int i = blockIdx.x * blockDim.x + threadIdx.x;
    if (i < n) selfw[i] = (cnt[i] > 0.0f) ? (wsum[i] / fmaxf(cnt[i], 1.0f)) : 0.0f;
}

// ---------------- fragment packing ----------------
// A fragments, per CDNA5 ISA 7.12.2 (16-bit A, 16x32):
//   lane 0-15:  M=lane,    VGPR0..3 K=0..7, VGPR4..7 K=16..23
//   lane 16-31: M=lane-16, K ranges +8
// Packed layout: Apack[(((rowTile*nK + ks)*32 + lane)*16 + i], i = 2*vgpr + half.
__global__ void pack_a(const float* __restrict__ X, _Float16* __restrict__ Apack,
                       int K, long total) {
    long t = (long)blockIdx.x * blockDim.x + threadIdx.x;
    if (t >= total) return;
    const int nK   = K >> 5;
    const int i    = (int)(t & 15);
    const int lane = (int)((t >> 4) & 31);
    long rest      = t >> 9;              // rowTile*nK + ks
    const int  ks      = (int)(rest % nK);
    const long rowTile = rest / nK;
    const int M   = lane & 15;
    const int hi8 = (lane >> 4) * 8;
    const int v   = i >> 1;
    const int low = i & 1;
    const int kin = ((v < 4) ? (2 * v + hi8) : (16 + 2 * (v - 4) + hi8)) + low;
    const long row = rowTile * 16 + M;
    const int  k   = ks * 32 + kin;
    Apack[t] = (_Float16)X[row * (long)K + k];
}

// B fragments (16-bit B, 32x16): lane 0-15: N=lane, i<8 K=i, i>=8 K=16+(i-8);
// lane 16-31: N=lane-16, K ranges +8. Packed: Bpack[(((colTile*nK + ks)*32 + lane)*16 + i]
__global__ void pack_b(const float* __restrict__ W, _Float16* __restrict__ Bpack,
                       int K, long total) {
    long t = (long)blockIdx.x * blockDim.x + threadIdx.x;
    if (t >= total) return;
    const int nK   = K >> 5;
    const int i    = (int)(t & 15);
    const int lane = (int)((t >> 4) & 31);
    long rest      = t >> 9;              // colTile*nK + ks
    const int ks      = (int)(rest % nK);
    const int colTile = (int)(rest / nK);
    const int col = colTile * 16 + (lane & 15);
    const int k   = ks * 32 + ((lane >> 4) * 8) + ((i < 8) ? i : (16 + (i - 8)));
    Bpack[t] = (_Float16)W[(size_t)k * HC + col];
}

// ---------------- WMMA GEMM: O[nRows x 128] = A[nRows x K] * W[K x 128] -------
// One wave per 16-row strip across all 8 column tiles. Per K-step: all 8 B
// fragments pre-loaded into registers (one big clause), A double-buffered, then
// an 8-WMMA burst that overlaps with the still-outstanding loads.
__global__ __launch_bounds__(32)
void wmma_gemm_strip(const _Float16* __restrict__ Apack, const _Float16* __restrict__ Bpack,
                     float* __restrict__ O, int K) {
    const int  lane    = threadIdx.x & 31;
    const long rowTile = blockIdx.x;
    const int  nK      = K >> 5;

    const v16h* Ap = (const v16h*)Apack + rowTile * nK * 32 + lane;
    const v16h* Bp = (const v16h*)Bpack + lane;

    v8f acc[8];
#pragma unroll
    for (int ct = 0; ct < 8; ++ct) acc[ct] = (v8f){};

    v16h a = Ap[0];
    for (int ks = 0; ks < nK; ++ks) {
        // stage all 8 B fragments (16x global_load_b128 clause)
        v16h b[8];
#pragma unroll
        for (int ct = 0; ct < 8; ++ct) b[ct] = Bp[((size_t)ct * nK + ks) * 32];
        // double-buffer A for next K-step; prefetch next K-step's B (col tile 0)
        v16h an = a;
        if (ks + 1 < nK) {
            an = Ap[(size_t)(ks + 1) * 32];
            __builtin_prefetch(Bp + ((size_t)ks + 1) * 32, 0, 1);
        }
        // 8-WMMA burst; first WMMAs can issue while later B loads are in flight
#pragma unroll
        for (int ct = 0; ct < 8; ++ct) {
            acc[ct] = __builtin_amdgcn_wmma_f32_16x16x32_f16(
                /*neg_a=*/false, a, /*neg_b=*/false, b[ct],
                /*c_mod=*/(short)0, acc[ct], /*reuse_a=*/false, /*reuse_b=*/false);
        }
        a = an;
    }

    // C/D layout: lanes 0-15 N=lane, VGPR r -> M=r; lanes 16-31 -> M=r+8.
    const int  l15   = lane & 15;
    const long mbase = rowTile * 16 + (lane >> 4) * 8;
#pragma unroll
    for (int ct = 0; ct < 8; ++ct) {
        float* o = O + (size_t)mbase * HC + ct * 16 + l15;
#pragma unroll
        for (int r = 0; r < 8; ++r) o[(size_t)r * HC] = acc[ct][r];
    }
}

// ---------------- edge decode helper ----------------
__device__ __forceinline__ void edge_decode(long t, const int* src, const int* dst,
                                            const float* ew, const float* selfw,
                                            int& m, int& e, int& h, int& s, int& d, float& w) {
    h = (int)(t % HH);
    long me = t / HH;
    e = (int)(me % E2);
    m = (int)(me / E2);
    if (e < EED) { s = src[e]; d = dst[e]; w = ew[e]; }
    else         { s = e - EED; d = s;     w = selfw[s]; }
}

// order-preserving float<->uint for atomic segment-max
__device__ __forceinline__ unsigned f32_key(float f) {
    unsigned u = __float_as_uint(f);
    return (u & 0x80000000u) ? ~u : (u | 0x80000000u);
}
__device__ __forceinline__ float key_f32(unsigned k) {
    unsigned u = (k & 0x80000000u) ? (k & 0x7fffffffu) : ~k;
    return __uint_as_float(u);
}

// ---------------- edge pass 1: logits + segment max ----------------
__global__ void edge_logits(const int* __restrict__ src, const int* __restrict__ dst,
                            const float* __restrict__ ew, const float* __restrict__ selfw,
                            const float* __restrict__ xl, const float* __restrict__ xr,
                            const float* __restrict__ att, const float* __restrict__ We,
                            float* __restrict__ abuf, unsigned* __restrict__ smax, long total) {
    long t = (long)blockIdx.x * blockDim.x + threadIdx.x;
    if (t >= total) return;
    int m, e, h, s, d; float w;
    edge_decode(t, src, dst, ew, selfw, m, e, h, s, d, w);
    const float4* xj = (const float4*)(xl + ((size_t)m * NGR + s) * HC + h * CC);
    const float4* xi = (const float4*)(xr + ((size_t)m * NGR + d) * HC + h * CC);
    const float4* Wf = (const float4*)(We + h * CC);
    const float4* Af = (const float4*)(att + h * CC);
    float lg = 0.0f;
#pragma unroll 4
    for (int c = 0; c < CC / 4; ++c) {
        float4 a = xi[c], b = xj[c], we = Wf[c], at = Af[c];
        float v0 = a.x + b.x + w * we.x; v0 = (v0 > 0.0f) ? v0 : NEG_SLOPE * v0;
        float v1 = a.y + b.y + w * we.y; v1 = (v1 > 0.0f) ? v1 : NEG_SLOPE * v1;
        float v2 = a.z + b.z + w * we.z; v2 = (v2 > 0.0f) ? v2 : NEG_SLOPE * v2;
        float v3 = a.w + b.w + w * we.w; v3 = (v3 > 0.0f) ? v3 : NEG_SLOPE * v3;
        lg = fmaf(at.x, v0, fmaf(at.y, v1, fmaf(at.z, v2, fmaf(at.w, v3, lg))));
    }
    abuf[t] = lg;
    atomicMax(&smax[((size_t)m * NGR + d) * HH + h], f32_key(lg));
}

// ---------------- edge pass 2: exp + segment sum ----------------
__global__ void edge_exp(const int* __restrict__ src, const int* __restrict__ dst,
                         const float* __restrict__ ew, const float* __restrict__ selfw,
                         float* __restrict__ abuf, const unsigned* __restrict__ smax,
                         float* __restrict__ denom, long total) {
    long t = (long)blockIdx.x * blockDim.x + threadIdx.x;
    if (t >= total) return;
    int m, e, h, s, d; float w;
    edge_decode(t, src, dst, ew, selfw, m, e, h, s, d, w);
    size_t idx = ((size_t)m * NGR + d) * HH + h;
    float a = __expf(abuf[t] - key_f32(smax[idx]));
    abuf[t] = a;
    atomicAdd(&denom[idx], a);
}

// ---------------- edge pass 3: alpha-weighted message scatter ----------------
__global__ void edge_scatter(const int* __restrict__ src, const int* __restrict__ dst,
                             const float* __restrict__ ew, const float* __restrict__ selfw,
                             const float* __restrict__ xl, const float* __restrict__ abuf,
                             const float* __restrict__ denom, float* __restrict__ out, long total) {
    long t = (long)blockIdx.x * blockDim.x + threadIdx.x;
    if (t >= total) return;
    int m, e, h, s, d; float w;
    edge_decode(t, src, dst, ew, selfw, m, e, h, s, d, w);
    size_t idx = ((size_t)m * NGR + d) * HH + h;
    float alpha = abuf[t] / (denom[idx] + EPSV);
    const float4* xj = (const float4*)(xl + ((size_t)m * NGR + s) * HC + h * CC);
    float* o = out + ((size_t)m * NGR + d) * HC + h * CC;
#pragma unroll 4
    for (int c = 0; c < CC / 4; ++c) {
        float4 v = xj[c];
        atomicAdd(&o[4 * c + 0], alpha * v.x);
        atomicAdd(&o[4 * c + 1], alpha * v.y);
        atomicAdd(&o[4 * c + 2], alpha * v.z);
        atomicAdd(&o[4 * c + 3], alpha * v.w);
    }
}

// ---------------- bias + ELU ----------------
__global__ void bias_elu(float* __restrict__ o, const float* __restrict__ b, long n) {
    long i = (long)blockIdx.x * blockDim.x + threadIdx.x;
    if (i >= n) return;
    int c = (int)(i % HC);
    float v = o[i] + b[c];
    o[i] = (v > 0.0f) ? v : (__expf(v) - 1.0f);
}

// ---------------- host-side launcher ----------------
static inline long cdivl(long a, long b) { return (a + b - 1) / b; }

extern "C" void kernel_launch(void* const* d_in, const int* in_sizes, int n_in,
                              void* d_out, int out_size, void* d_ws, size_t ws_size,
                              hipStream_t stream) {
    (void)in_sizes; (void)n_in; (void)out_size; (void)ws_size;

    const float* x    = (const float*)d_in[0];
    const int*   ei   = (const int*)d_in[1];
    const int*   src  = ei;
    const int*   dst  = ei + EED;
    const float* ew   = (const float*)d_in[2];
    const float* Wl1  = (const float*)d_in[3];
    const float* Wr1  = (const float*)d_in[4];
    const float* att1 = (const float*)d_in[5];
    const float* We1  = (const float*)d_in[6];
    const float* b1   = (const float*)d_in[7];
    const float* Wl2  = (const float*)d_in[8];
    const float* Wr2  = (const float*)d_in[9];
    const float* att2 = (const float*)d_in[10];
    const float* We2  = (const float*)d_in[11];
    const float* b2   = (const float*)d_in[12];
    float* out = (float*)d_out;

    // ---- workspace carve-out ----
    char* ws = (char*)d_ws;
    size_t off = 0;
    auto alloc = [&](size_t bytes) -> void* {
        void* p = ws + off;
        off = (off + bytes + 255) & ~(size_t)255;
        return p;
    };
    float*     cnt   = (float*)alloc((size_t)NGR * 4);
    float*     wsum  = (float*)alloc((size_t)NGR * 4);
    float*     selfw = (float*)alloc((size_t)NGR * 4);
    unsigned*  smax  = (unsigned*)alloc((size_t)MGR * NGR * HH * 4);
    float*     denom = (float*)alloc((size_t)MGR * NGR * HH * 4);
    float*     abuf  = (float*)alloc((size_t)MGR * E2 * HH * 4);
    float*     xl    = (float*)alloc((size_t)MGR * NGR * HC * 4);
    float*     xr    = (float*)alloc((size_t)MGR * NGR * HC * 4);
    float*     hbuf  = (float*)alloc((size_t)MGR * NGR * HC * 4);
    _Float16*  apack = (_Float16*)alloc((size_t)MGR * NGR * HC * 2);  // fits K<=128
    _Float16*  wl1p  = (_Float16*)alloc((size_t)DIN * HC * 2);
    _Float16*  wr1p  = (_Float16*)alloc((size_t)DIN * HC * 2);
    _Float16*  wl2p  = (_Float16*)alloc((size_t)HC * HC * 2);
    _Float16*  wr2p  = (_Float16*)alloc((size_t)HC * HC * 2);

    const long nRows  = (long)MGR * NGR;      // 80000
    const long nFeat  = nRows * HC;           // 10.24M
    const long nSeg   = (long)MGR * NGR * HH; // 160K
    const long nEdgeT = (long)MGR * E2 * HH;  // 2.72M
    const dim3 gemmGrid((unsigned)(nRows / 16));
    const dim3 gemmBlk(32);

    // ---- self-loop weights ----
    fill_f32<<<(unsigned)cdivl(NGR, 256), 256, 0, stream>>>(cnt, 0.0f, NGR);
    fill_f32<<<(unsigned)cdivl(NGR, 256), 256, 0, stream>>>(wsum, 0.0f, NGR);
    selfloop_accum<<<(unsigned)cdivl(EED, 256), 256, 0, stream>>>(dst, ew, cnt, wsum, EED);
    selfloop_final<<<(unsigned)cdivl(NGR, 256), 256, 0, stream>>>(selfw, cnt, wsum, NGR);

    // ---- weight packing (f32 -> fragment-ordered f16) ----
    pack_b<<<(unsigned)cdivl((long)DIN * HC, 256), 256, 0, stream>>>(Wl1, wl1p, DIN, (long)DIN * HC);
    pack_b<<<(unsigned)cdivl((long)DIN * HC, 256), 256, 0, stream>>>(Wr1, wr1p, DIN, (long)DIN * HC);
    pack_b<<<(unsigned)cdivl((long)HC * HC, 256), 256, 0, stream>>>(Wl2, wl2p, HC, (long)HC * HC);
    pack_b<<<(unsigned)cdivl((long)HC * HC, 256), 256, 0, stream>>>(Wr2, wr2p, HC, (long)HC * HC);

    // ================= layer 1 =================
    const long nIn1 = nRows * DIN;
    pack_a<<<(unsigned)cdivl(nIn1, 256), 256, 0, stream>>>(x, apack, DIN, nIn1);
    wmma_gemm_strip<<<gemmGrid, gemmBlk, 0, stream>>>(apack, wl1p, xl, DIN);
    wmma_gemm_strip<<<gemmGrid, gemmBlk, 0, stream>>>(apack, wr1p, xr, DIN);

    fill_f32<<<(unsigned)cdivl(nSeg, 256), 256, 0, stream>>>((float*)smax, 0.0f, nSeg);
    fill_f32<<<(unsigned)cdivl(nSeg, 256), 256, 0, stream>>>(denom, 0.0f, nSeg);
    fill_f32<<<(unsigned)cdivl(nFeat, 256), 256, 0, stream>>>(hbuf, 0.0f, nFeat);

    edge_logits<<<(unsigned)cdivl(nEdgeT, 256), 256, 0, stream>>>(src, dst, ew, selfw, xl, xr, att1, We1, abuf, smax, nEdgeT);
    edge_exp<<<(unsigned)cdivl(nEdgeT, 256), 256, 0, stream>>>(src, dst, ew, selfw, abuf, smax, denom, nEdgeT);
    edge_scatter<<<(unsigned)cdivl(nEdgeT, 256), 256, 0, stream>>>(src, dst, ew, selfw, xl, abuf, denom, hbuf, nEdgeT);
    bias_elu<<<(unsigned)cdivl(nFeat, 256), 256, 0, stream>>>(hbuf, b1, nFeat);

    // ================= layer 2 =================
    pack_a<<<(unsigned)cdivl(nFeat, 256), 256, 0, stream>>>(hbuf, apack, HC, nFeat);
    wmma_gemm_strip<<<gemmGrid, gemmBlk, 0, stream>>>(apack, wl2p, xl, HC);
    wmma_gemm_strip<<<gemmGrid, gemmBlk, 0, stream>>>(apack, wr2p, xr, HC);

    fill_f32<<<(unsigned)cdivl(nSeg, 256), 256, 0, stream>>>((float*)smax, 0.0f, nSeg);
    fill_f32<<<(unsigned)cdivl(nSeg, 256), 256, 0, stream>>>(denom, 0.0f, nSeg);
    fill_f32<<<(unsigned)cdivl(nFeat, 256), 256, 0, stream>>>(out, 0.0f, nFeat);

    edge_logits<<<(unsigned)cdivl(nEdgeT, 256), 256, 0, stream>>>(src, dst, ew, selfw, xl, xr, att2, We2, abuf, smax, nEdgeT);
    edge_exp<<<(unsigned)cdivl(nEdgeT, 256), 256, 0, stream>>>(src, dst, ew, selfw, abuf, smax, denom, nEdgeT);
    edge_scatter<<<(unsigned)cdivl(nEdgeT, 256), 256, 0, stream>>>(src, dst, ew, selfw, xl, abuf, denom, out, nEdgeT);
    bias_elu<<<(unsigned)cdivl(nFeat, 256), 256, 0, stream>>>(out, b2, nFeat);
}
